// SelectiveStateSpaceLayer_87677462380892
// MI455X (gfx1250) — compile-verified, matrix operand order checked by reading, not run on hardware
//
#include <hip/hip_runtime.h>
#include <hip/hip_bf16.h>

// ---- fixed reference shapes ----
#define Bsz 4
#define Lseq 4096
#define Dch 512
#define Nst 64
#define Rrk 16
#define RN2 (Rrk + 2 * Nst)          // 144
#define BT (Bsz * Lseq)              // 16384
#define BIG_CLAMP 50.0f
#define CLAMP6 1.0e6f
#define EPS_F 1.0e-8f

typedef __attribute__((ext_vector_type(16))) __bf16 v16bf;
typedef __attribute__((ext_vector_type(8)))  float  v8f;

// K offset within a 32-wide K-chunk for vector pair index v (0..7), per wave32 WMMA A/B layout.
// lanes 0-15 (grp 0): K pairs {0,2,4,6,16,18,20,22}; lanes 16-31 (grp 1): +8
__device__ __forceinline__ int wmma_koff(int grp, int v) {
    return grp * 8 + ((v < 4) ? (2 * v) : (16 + 2 * (v - 4)));
}

// ---------------- K0: Aneg[d,n] = -exp(min(A_log[d,n], 50)) ----------------
__global__ void k_Aneg(const float* __restrict__ A_log, float* __restrict__ Aneg) {
    int i = blockIdx.x * 256 + threadIdx.x;
    if (i < Dch * Nst) {
        float v = A_log[i];
        v = fminf(v, BIG_CLAMP);
        Aneg[i] = -__expf(v);
    }
}

// ---------------- K1: xdbl = x @ Wx + bx  (WMMA bf16, f32 acc) ----------------
// One wave computes a 16(M) x 144(N) stripe: A loaded+converted once per K-chunk,
// 9 independent WMMAs back-to-back per chunk. x read exactly once from HBM.
__global__ __launch_bounds__(32) void k_gemm_xdbl(const float* __restrict__ x,
                                                  const float* __restrict__ Wx,
                                                  const float* __restrict__ bx,
                                                  float* __restrict__ xdbl) {
    const int NT = RN2 / 16;                 // 9 col tiles
    int mt   = blockIdx.x;                   // 0..1023
    int lane = threadIdx.x;
    int grp  = lane >> 4;                    // 0 or 1
    int lid  = lane & 15;
    int arow = mt * 16 + lid;                // A row for this lane

    float bias[NT];
#pragma unroll
    for (int nt = 0; nt < NT; ++nt) bias[nt] = bx[nt * 16 + lid];

    v8f acc[NT];
#pragma unroll
    for (int nt = 0; nt < NT; ++nt) acc[nt] = (v8f){};

    for (int k0 = 0; k0 < Dch; k0 += 32) {
        v16bf av;
#pragma unroll
        for (int v = 0; v < 8; ++v) {
            int k = k0 + wmma_koff(grp, v);
            float a0 = x[arow * Dch + k];
            float a1 = x[arow * Dch + k + 1];
            av[2 * v]     = (__bf16)a0;
            av[2 * v + 1] = (__bf16)a1;
        }
#pragma unroll
        for (int nt = 0; nt < NT; ++nt) {
            int bcol = nt * 16 + lid;
            v16bf bv;
#pragma unroll
            for (int v = 0; v < 8; ++v) {
                int k = k0 + wmma_koff(grp, v);
                bv[2 * v]     = (__bf16)Wx[k * RN2 + bcol];
                bv[2 * v + 1] = (__bf16)Wx[(k + 1) * RN2 + bcol];
            }
            acc[nt] = __builtin_amdgcn_wmma_f32_16x16x32_bf16(false, av, false, bv,
                                                              (short)0, acc[nt],
                                                              false, false);
        }
    }
    // D layout: VGPR r -> M = mt*16 + grp*8 + r, N = nt*16 + lid
    int mbase = mt * 16 + grp * 8;
#pragma unroll
    for (int nt = 0; nt < NT; ++nt) {
#pragma unroll
        for (int r = 0; r < 8; ++r) {
            xdbl[(mbase + r) * RN2 + nt * 16 + lid] = acc[nt][r] + bias[nt];
        }
    }
}

// ---------------- K2: dt = softplus(xdbl[:, :16] @ Wdt + bdt) + eps ----------------
// One wave handles 8 N-tiles: A tile (K=16, zero-padded to 32) converted once, 8 WMMAs.
__global__ __launch_bounds__(32) void k_gemm_dt(const float* __restrict__ xdbl,
                                                 const float* __restrict__ Wdt,
                                                 const float* __restrict__ bdt,
                                                 float* __restrict__ dtb) {
    const int NTW = 8;                       // N-tiles per wave
    int mt   = blockIdx.x >> 2;              // 1024 row tiles
    int ntg  = blockIdx.x & 3;               // 4 groups of 8 tiles (32 total)
    int lane = threadIdx.x;
    int grp  = lane >> 4;
    int lid  = lane & 15;
    int arow = mt * 16 + lid;

    v16bf av;
#pragma unroll
    for (int v = 0; v < 8; ++v) {
        int k = wmma_koff(grp, v);
        float a0 = (k < Rrk)     ? xdbl[arow * RN2 + k]     : 0.0f;
        float a1 = (k + 1 < Rrk) ? xdbl[arow * RN2 + k + 1] : 0.0f;
        av[2 * v]     = (__bf16)a0;
        av[2 * v + 1] = (__bf16)a1;
    }

    int mbase = mt * 16 + grp * 8;
#pragma unroll
    for (int j = 0; j < NTW; ++j) {
        int nt   = ntg * NTW + j;
        int bcol = nt * 16 + lid;
        v16bf bv;
#pragma unroll
        for (int v = 0; v < 8; ++v) {
            int k = wmma_koff(grp, v);
            bv[2 * v]     = (__bf16)((k < Rrk)     ? Wdt[k * Dch + bcol]       : 0.0f);
            bv[2 * v + 1] = (__bf16)((k + 1 < Rrk) ? Wdt[(k + 1) * Dch + bcol] : 0.0f);
        }
        v8f acc = {};
        acc = __builtin_amdgcn_wmma_f32_16x16x32_bf16(false, av, false, bv,
                                                      (short)0, acc, false, false);
        float bias = bdt[bcol];
#pragma unroll
        for (int r = 0; r < 8; ++r) {
            float z = acc[r] + bias;
            float sp = (z > 20.0f) ? z : log1pf(__expf(z));   // softplus
            dtb[(mbase + r) * Dch + bcol] = sp + EPS_F;
        }
    }
}

// ---------------- K3: per-(b,t) coefficients (exp-reduction over D) ----------------
// a[bt,n] = clip( sum_d exp(min(Aneg[d,n]*dt[bt,d], 50)) )
// b[bt,n] = sum_d clip( Bm[bt,n]*dt[bt,d]*x[bt,d] )
// Each block handles GBT=8 timesteps; Aneg staged in LDS chunk-wise and shared
// across the 8 accumulator sets -> 8x less global Aneg traffic.
#define GBT 8
#define DCHUNK 128
__global__ __launch_bounds__(64) void k_coeff(const float* __restrict__ x,
                                              const float* __restrict__ xdbl,
                                              const float* __restrict__ dtb,
                                              const float* __restrict__ Aneg,
                                              float* __restrict__ abuf,
                                              float* __restrict__ bbuf) {
    __shared__ float sA[DCHUNK * Nst];       // 32 KB
    __shared__ float sdt[GBT][Dch];          // 16 KB
    __shared__ float sx[GBT][Dch];           // 16 KB
    int bt0 = blockIdx.x * GBT;
    int n   = threadIdx.x;                   // 0..63

#pragma unroll
    for (int g = 0; g < GBT; ++g) {
        for (int d = n; d < Dch; d += 64) {
            sdt[g][d] = dtb[(bt0 + g) * Dch + d];
            sx[g][d]  = x[(bt0 + g) * Dch + d];
        }
    }

    float Bmn[GBT], suma[GBT], sumb[GBT];
#pragma unroll
    for (int g = 0; g < GBT; ++g) {
        Bmn[g]  = xdbl[(bt0 + g) * RN2 + Rrk + n];
        suma[g] = 0.0f;
        sumb[g] = 0.0f;
    }

    for (int dc = 0; dc < Dch; dc += DCHUNK) {
        __syncthreads();
        for (int i = n; i < DCHUNK * Nst; i += 64) sA[i] = Aneg[dc * Nst + i];
        __syncthreads();
#pragma unroll 2
        for (int d = 0; d < DCHUNK; ++d) {
            float An = sA[d * Nst + n];      // bank-conflict-free
#pragma unroll
            for (int g = 0; g < GBT; ++g) {
                float dtv = sdt[g][dc + d];  // LDS broadcast
                suma[g] += __expf(fminf(An * dtv, BIG_CLAMP));
                float t2 = Bmn[g] * dtv * sx[g][dc + d];
                sumb[g] += fminf(fmaxf(t2, -CLAMP6), CLAMP6);
            }
        }
    }
#pragma unroll
    for (int g = 0; g < GBT; ++g) {
        float sa = fminf(fmaxf(suma[g], -CLAMP6), CLAMP6);
        abuf[(bt0 + g) * Nst + n] = sa;
        bbuf[(bt0 + g) * Nst + n] = sumb[g];
    }
}

// ---------------- K4: sequential scan h = a*h + b, fused s[b,t] = sum_n C*h ----------------
// one wave32 per batch; each lane owns n and n+32; wave-reduce for the N-sum.
__global__ __launch_bounds__(32) void k_scan(const float* __restrict__ abuf,
                                             const float* __restrict__ bbuf,
                                             const float* __restrict__ xdbl,
                                             float* __restrict__ sbuf) {
    int b    = blockIdx.x;
    int lane = threadIdx.x;
    int n0 = lane, n1 = lane + 32;
    float h0 = 0.0f, h1 = 0.0f;
    for (int t = 0; t < Lseq; ++t) {
        int idx = b * Lseq + t;
        // prefetch future iterations to keep the dependent chain at issue rate
        if (t + 16 < Lseq) {
            int pidx = idx + 16;
            __builtin_prefetch(&abuf[pidx * Nst + n0], 0, 0);
            __builtin_prefetch(&bbuf[pidx * Nst + n0], 0, 0);
            __builtin_prefetch(&xdbl[pidx * RN2 + Rrk + Nst + n0], 0, 0);
        }
        float a0 = abuf[idx * Nst + n0];
        float a1 = abuf[idx * Nst + n1];
        float v0 = bbuf[idx * Nst + n0];
        float v1 = bbuf[idx * Nst + n1];
        h0 = a0 * h0 + v0;
        h1 = a1 * h1 + v1;
        float C0 = xdbl[idx * RN2 + Rrk + Nst + n0];
        float C1 = xdbl[idx * RN2 + Rrk + Nst + n1];
        float p = C0 * h0 + C1 * h1;
#pragma unroll
        for (int off = 16; off > 0; off >>= 1)
            p += __shfl_down(p, off, 32);
        if (lane == 0) sbuf[idx] = p;
    }
}

// ---------------- K5: y[b,t,d] = s[b,t] + Dp[d]*x[b,t,d] ----------------
__global__ void k_out(const float* __restrict__ x,
                      const float* __restrict__ sbuf,
                      const float* __restrict__ Dp,
                      float* __restrict__ y) {
    int i = blockIdx.x * 256 + threadIdx.x;
    if (i < BT * Dch) {
        int bt = i >> 9;        // / Dch
        int d  = i & (Dch - 1);
        y[i] = sbuf[bt] + Dp[d] * x[i];
    }
}

// ---------------- launcher ----------------
extern "C" void kernel_launch(void* const* d_in, const int* in_sizes, int n_in,
                              void* d_out, int out_size, void* d_ws, size_t ws_size,
                              hipStream_t stream) {
    const float* x     = (const float*)d_in[0];  // [B,L,D]
    const float* Wx    = (const float*)d_in[1];  // [D, R+2N]
    const float* bx    = (const float*)d_in[2];  // [R+2N]
    const float* Wdt   = (const float*)d_in[3];  // [R, D]
    const float* bdt   = (const float*)d_in[4];  // [D]
    const float* A_log = (const float*)d_in[5];  // [D, N]
    const float* Dp    = (const float*)d_in[6];  // [D]
    float* y = (float*)d_out;                    // [B,L,D]

    float* ws = (float*)d_ws;
    // workspace layout (floats)
    float* xdbl = ws;                                  // BT*RN2   = 2,359,296
    float* dtb  = xdbl + (size_t)BT * RN2;             // BT*Dch   = 8,388,608
    float* abuf = dtb  + (size_t)BT * Dch;             // BT*Nst   = 1,048,576
    float* bbuf = abuf + (size_t)BT * Nst;             // BT*Nst   = 1,048,576
    float* Aneg = bbuf + (size_t)BT * Nst;             // Dch*Nst  = 32,768
    float* sbuf = Aneg + (size_t)Dch * Nst;            // BT       = 16,384

    // K0: Aneg
    k_Aneg<<<(Dch * Nst + 255) / 256, 256, 0, stream>>>(A_log, Aneg);
    // K1: xdbl = x @ Wx + bx   (WMMA, 16x144 stripe per wave)
    k_gemm_xdbl<<<BT / 16, 32, 0, stream>>>(x, Wx, bx, xdbl);
    // K2: dt = softplus(xdbl[:, :16] @ Wdt + bdt) + eps   (WMMA, 8 tiles per wave)
    k_gemm_dt<<<(BT / 16) * 4, 32, 0, stream>>>(xdbl, Wdt, bdt, dtb);
    // K3: per-step coefficients (exp-heavy, 8 timesteps per block, Aneg in LDS)
    k_coeff<<<BT / GBT, 64, 0, stream>>>(x, xdbl, dtb, Aneg, abuf, bbuf);
    // K4: sequential scan + output reduction over N
    k_scan<<<Bsz, 32, 0, stream>>>(abuf, bbuf, xdbl, sbuf);
    // K5: final elementwise output
    k_out<<<(BT * Dch + 255) / 256, 256, 0, stream>>>(x, sbuf, Dp, y);
}
